// GATLayer1_13219909337125
// MI455X (gfx1250) — matrix operand order repaired and checked by reading, not compiled, not run on hardware
//
#include <hip/hip_runtime.h>
#include <hip/hip_bf16.h>

#define GN   8192
#define IN_F 512
#define OUT_F 64
#define ALPHA 0.2f
#define NEG_INF_F (-9e15f)
#define KWAVES 8                      // split-K waves per row-tile block

typedef __attribute__((ext_vector_type(16))) _Float16 v16h;
typedef __attribute__((ext_vector_type(8)))  float    v8f;
typedef __attribute__((ext_vector_type(4)))  float    v4f;

// ---------------- Kernel 1: Wh = x @ W  (8192x512 * 512x64) ----------------
__global__ __launch_bounds__(256) void k_xw(const float* __restrict__ x,
                                            const float* __restrict__ W,
                                            float* __restrict__ Wh) {
  __shared__ float xs[4][IN_F];                       // 8 KB
  const int r0 = blockIdx.x * 4;
  const int t  = threadIdx.x;
  for (int u = t; u < 4 * IN_F; u += 256)
    xs[u >> 9][u & 511] = x[(size_t)(r0 + (u >> 9)) * IN_F + (u & 511)];
  __syncthreads();
  const int f  = t & 63;
  const int jr = t >> 6;
  float acc = 0.f;
  #pragma unroll 8
  for (int k = 0; k < IN_F; ++k)
    acc = fmaf(xs[jr][k], W[k * OUT_F + f], acc);
  Wh[(size_t)(r0 + jr) * OUT_F + f] = acc;
}

// -- Kernel 2: s1 = Wh@a[:64], s2 = Wh@a[64:], Wh16T[f][k] = f16(Wh[k][f]) --
__global__ __launch_bounds__(256) void k_logits(const float* __restrict__ Wh,
                                                const float* __restrict__ a,
                                                _Float16* __restrict__ Wh16T,
                                                float* __restrict__ s1,
                                                float* __restrict__ s2) {
  const int row  = (blockIdx.x * 256 + threadIdx.x) >> 5;   // 8192 warps total
  const int lane = threadIdx.x & 31;
  const float w0 = Wh[(size_t)row * OUT_F + lane];
  const float w1 = Wh[(size_t)row * OUT_F + lane + 32];
  Wh16T[(size_t)lane * GN + row]        = (_Float16)w0;     // transposed copy
  Wh16T[(size_t)(lane + 32) * GN + row] = (_Float16)w1;
  float p1 = w0 * a[lane]      + w1 * a[lane + 32];
  float p2 = w0 * a[64 + lane] + w1 * a[96 + lane];
  #pragma unroll
  for (int off = 16; off; off >>= 1) {
    p1 += __shfl_down(p1, off, 32);
    p2 += __shfl_down(p2, off, 32);
  }
  if (lane == 0) { s1[row] = p1; s2[row] = p2; }
}

// -------- Kernel 3: masked leaky-relu logits + row softmax -> attention -----
// One block per row i; 32KB LDS row buffer so adj is read exactly once.
// adj / attention are pure streams -> non-temporal so L2 keeps Wh16T hot.
__global__ __launch_bounds__(256) void k_softmax(const int* __restrict__ adj,
                                                 const float* __restrict__ s1,
                                                 const float* __restrict__ s2,
                                                 float* __restrict__ attn) {
  __shared__ float rowb[GN];                          // 32 KB
  __shared__ float red[8];
  const int i = blockIdx.x;
  const int t = threadIdx.x;
  const int lane = t & 31, warp = t >> 5;
  const float s1i = s1[i];

  float m = NEG_INF_F;
  for (int j = t; j < GN; j += 256) {
    float e = s1i + s2[j];
    e = (e >= 0.f) ? e : ALPHA * e;                   // LeakyReLU
    const int aj = __builtin_nontemporal_load(adj + (size_t)i * GN + j);
    float v = (aj > 0) ? e : NEG_INF_F;
    rowb[j] = v;
    m = fmaxf(m, v);
  }
  #pragma unroll
  for (int off = 16; off; off >>= 1) m = fmaxf(m, __shfl_xor(m, off, 32));
  if (lane == 0) red[warp] = m;
  __syncthreads();
  float bm = red[0];
  #pragma unroll
  for (int q = 1; q < 8; ++q) bm = fmaxf(bm, red[q]);

  float sum = 0.f;
  for (int j = t; j < GN; j += 256) {
    float p = __expf(rowb[j] - bm);
    rowb[j] = p;
    sum += p;
  }
  #pragma unroll
  for (int off = 16; off; off >>= 1) sum += __shfl_xor(sum, off, 32);
  __syncthreads();                                    // all done reading red(max)
  if (lane == 0) red[warp] = sum;
  __syncthreads();
  float bs = 0.f;
  #pragma unroll
  for (int q = 0; q < 8; ++q) bs += red[q];
  const float inv = 1.f / bs;

  for (int j = t; j < GN; j += 256)
    __builtin_nontemporal_store(rowb[j] * inv, attn + (size_t)i * GN + j);
}

// ------ Kernel 4: h_prime = attention @ Wh via v_wmma_f32_16x16x32_f16 ------
// Block = 256 threads = 8 waves. Each block owns 16 output rows; wave w
// accumulates the full 16x64 tile over K-slice [w*1024, (w+1)*1024).
// No LDS / no barriers in the K loop: A comes from global (NT, f32->pk f16),
// B from the L2-resident transposed Wh16T. Split-K reduced through LDS + ELU.
__global__ __launch_bounds__(256) void k_attn_wh(const float* __restrict__ attn,
                                                 const _Float16* __restrict__ Wh16T,
                                                 float* __restrict__ out) {
  __shared__ float part[KWAVES][16][OUT_F];           // 32 KB
  const int i0    = blockIdx.x * 16;
  const int t     = threadIdx.x;
  const int wave  = t >> 5;
  const int lane  = t & 31;
  const int m     = lane & 15;                        // fragment row / col
  const int khalf = lane >> 4;                        // 0 or 1
  const int kbeg  = wave * (GN / KWAVES);
  const int kend  = kbeg + (GN / KWAVES);

  v8f c0 = {}, c1 = {}, c2 = {}, c3 = {};
  const float*    arow = attn  + (size_t)(i0 + m) * GN;
  const _Float16* brow = Wh16T + (size_t)m * GN;

  for (int k0 = kbeg; k0 < kend; k0 += 32) {
    // ---- A fragment: rows i0+m, K = khalf*16+{0..15}
    const float* ap = arow + k0 + khalf * 16;
    const v4f f0 = __builtin_nontemporal_load((const v4f*)(ap));
    const v4f f1 = __builtin_nontemporal_load((const v4f*)(ap + 4));
    const v4f f2 = __builtin_nontemporal_load((const v4f*)(ap + 8));
    const v4f f3 = __builtin_nontemporal_load((const v4f*)(ap + 12));
    v16h a;
    {
      auto h0 = __builtin_amdgcn_cvt_pkrtz(f0[0], f0[1]);
      auto h1 = __builtin_amdgcn_cvt_pkrtz(f0[2], f0[3]);
      auto h2 = __builtin_amdgcn_cvt_pkrtz(f1[0], f1[1]);
      auto h3 = __builtin_amdgcn_cvt_pkrtz(f1[2], f1[3]);
      auto h4 = __builtin_amdgcn_cvt_pkrtz(f2[0], f2[1]);
      auto h5 = __builtin_amdgcn_cvt_pkrtz(f2[2], f2[3]);
      auto h6 = __builtin_amdgcn_cvt_pkrtz(f3[0], f3[1]);
      auto h7 = __builtin_amdgcn_cvt_pkrtz(f3[2], f3[3]);
      a[0]=h0[0];  a[1]=h0[1];  a[2]=h1[0];  a[3]=h1[1];
      a[4]=h2[0];  a[5]=h2[1];  a[6]=h3[0];  a[7]=h3[1];
      a[8]=h4[0];  a[9]=h4[1];  a[10]=h5[0]; a[11]=h5[1];
      a[12]=h6[0]; a[13]=h6[1]; a[14]=h7[0]; a[15]=h7[1];
    }
    // ---- B fragments: col = ct*16+m, K = khalf*16+{0..15} (same k order as A)
    const _Float16* bp = brow + k0 + khalf * 16;
    const v16h b0 = *(const v16h*)(bp);
    const v16h b1 = *(const v16h*)(bp + (size_t)16 * GN);
    const v16h b2 = *(const v16h*)(bp + (size_t)32 * GN);
    const v16h b3 = *(const v16h*)(bp + (size_t)48 * GN);

    c0 = __builtin_amdgcn_wmma_f32_16x16x32_f16(false, a, false, b0, (short)0, c0, false, false);
    c1 = __builtin_amdgcn_wmma_f32_16x16x32_f16(false, a, false, b1, (short)0, c1, false, false);
    c2 = __builtin_amdgcn_wmma_f32_16x16x32_f16(false, a, false, b2, (short)0, c2, false, false);
    c3 = __builtin_amdgcn_wmma_f32_16x16x32_f16(false, a, false, b3, (short)0, c3, false, false);
  }

  // ---- split-K reduction: D lane layout -> LDS, then coalesced ELU store
  #pragma unroll
  for (int r = 0; r < 8; ++r) {
    const int rr = r + 8 * khalf;
    part[wave][rr][ 0 + m] = c0[r];
    part[wave][rr][16 + m] = c1[r];
    part[wave][rr][32 + m] = c2[r];
    part[wave][rr][48 + m] = c3[r];
  }
  __syncthreads();
  #pragma unroll
  for (int e = t; e < 16 * OUT_F; e += 256) {
    const int rr = e >> 6, cc = e & 63;
    float v = 0.f;
    #pragma unroll
    for (int w = 0; w < KWAVES; ++w) v += part[w][rr][cc];
    v = (v > 0.f) ? v : expm1f(v);                    // ELU
    out[(size_t)(i0 + rr) * OUT_F + cc] = v;
  }
}

extern "C" void kernel_launch(void* const* d_in, const int* in_sizes, int n_in,
                              void* d_out, int out_size, void* d_ws, size_t ws_size,
                              hipStream_t stream) {
  const float* x   = (const float*)d_in[0];
  const int*   adj = (const int*)d_in[1];
  const float* W   = (const float*)d_in[2];
  const float* a   = (const float*)d_in[3];

  float* out  = (float*)d_out;                        // [8192, 64]
  float* attn = out + (size_t)GN * OUT_F;             // [8192, 8192]

  char* ws = (char*)d_ws;                             // ~3.1 MB used
  float*    Wh    = (float*)ws;                                      // 2 MB
  _Float16* Wh16T = (_Float16*)(ws + (size_t)GN * OUT_F * 4);        // 1 MB
  float*    s1    = (float*)(ws + (size_t)GN * OUT_F * 6);           // 32 KB
  float*    s2    = s1 + GN;                                         // 32 KB

  k_xw     <<<GN / 4,  256, 0, stream>>>(x, W, Wh);
  k_logits <<<GN / 8,  256, 0, stream>>>(Wh, a, Wh16T, s1, s2);
  k_softmax<<<GN,      256, 0, stream>>>(adj, s1, s2, attn);
  k_attn_wh<<<GN / 16, 256, 0, stream>>>(attn, Wh16T, out);
}
